// Model13_9620726743226
// MI455X (gfx1250) — matrix operand order repaired and checked by reading, not compiled, not run on hardware
//
#include <hip/hip_runtime.h>
#include <hip/hip_bf16.h>
#include <math.h>

typedef __attribute__((ext_vector_type(16))) _Float16 v16h;
typedef __attribute__((ext_vector_type(8)))  float    v8f;

#define NTOT   65536      // total nodes (B*N)
#define NB     512        // graphs
#define NPG    128        // nodes per graph
#define EDGES  1048576    // 2*E_HALF directed edges
#define AORD   131072
#define TORD   131072
#define DORD   65536
#define NMOV   16
#define NSEG   (NB*NMOV)  // 8192

// =====================================================================
// Fragment-ordered operand layouts (we own the packed formats):
//
// A tile (16 rows x KPAD cols f16) in LDS, frag order:
//   element (r, c): chunk = c/32, kk = c%32
//     h = kk >> 4               (K-half within lane group)
//     j = kk & 15               (position within the 16 halves)
//     lane = r + 16*h
//   offset = chunk*512 + lane*16 + j      (f16 units)
//   -> per chunk, lane L reads 16 contiguous f16 = one aligned 32B load.
//
// B weight ([KPAD x NOUT] f16), frag order:
//   element (k, n): chunk = k/32, kk = k%32, ntile = n/16
//     lane = (n & 15) + 16*(kk>>4)
//     j    = kk & 15
//   offset = ((chunk*tilesN + ntile)*32 + lane)*16 + j
//   -> per (chunk, ntile), lane L reads 16 contiguous f16 = 32B load.
// =====================================================================

__device__ inline void atomicMaxF(float* addr, float val) {
  if (val >= 0.f) atomicMax((int*)addr, __float_as_int(val));
  else            atomicMin((unsigned int*)addr, (unsigned int)__float_as_int(val));
}

__device__ inline int afrag_off(int r, int c) {
  int chunk = c >> 5, kk = c & 31;
  int h = kk >> 4;
  int j = kk & 15;
  return chunk * 512 + (r + 16 * h) * 16 + j;
}

// ======================= utility / pack kernels =======================
__global__ void fill_f32(float* p, int n, float v) {
  int i = blockIdx.x * blockDim.x + threadIdx.x;
  if (i < n) p[i] = v;
}

// pack 4 weight mats [K x 64] f32 -> frag-ordered [kpad x 256] f16
__global__ void pack_w4(const float* W0, const float* W1, const float* W2, const float* W3,
                        int K, int kpad, _Float16* out) {
  int idx = blockIdx.x * blockDim.x + threadIdx.x;
  int total = kpad * 256;
  if (idx >= total) return;
  int k = idx >> 8, col = idx & 255;
  int s = col >> 6, c = col & 63;
  const float* W = (s == 0) ? W0 : (s == 1) ? W1 : (s == 2) ? W2 : W3;
  float v = (k < K) ? W[k * 64 + c] : 0.f;
  int chunk = k >> 5, kk = k & 31, ntile = col >> 4;
  int lane = (col & 15) + 16 * (kk >> 4);
  int j = kk & 15;
  out[((chunk * 16 + ntile) * 32 + lane) * 16 + j] = (_Float16)v;  // tilesN=16
}

// pack one [K x 64] f32 -> frag-ordered [kpad x 64] f16 (tilesN = 4)
__global__ void pack_w1(const float* W, int K, int kpad, _Float16* out) {
  int idx = blockIdx.x * blockDim.x + threadIdx.x;
  int total = kpad * 64;
  if (idx >= total) return;
  int k = idx >> 6, col = idx & 63;
  float v = (k < K) ? W[k * 64 + col] : 0.f;
  int chunk = k >> 5, kk = k & 31, ntile = col >> 4;
  int lane = (col & 15) + 16 * (kk >> 4);
  int j = kk & 15;
  out[((chunk * 4 + ntile) * 32 + lane) * 16 + j] = (_Float16)v;
}

__global__ void pack_b4(const float* b0, const float* b1, const float* b2, const float* b3,
                        float* out) {
  int i = threadIdx.x;  // 256 threads
  const float* b = (i < 64) ? b0 : (i < 128) ? b1 : (i < 192) ? b2 : b3;
  out[i] = b[i & 63];
}

// fold: out = (W @ Wacc) frag-ordered f16 [kpad x 64] (tilesN = 4)
__global__ void fold_w(const float* W, const float* Wacc, int K, int kpad, _Float16* out) {
  int idx = blockIdx.x * blockDim.x + threadIdx.x;
  int total = kpad * 64;
  if (idx >= total) return;
  int k = idx >> 6, n = idx & 63;
  float acc = 0.f;
  if (k < K) {
#pragma unroll 8
    for (int j = 0; j < 64; ++j) acc += W[k * 64 + j] * Wacc[j * 64 + n];
  }
  int chunk = k >> 5, kk = k & 31, ntile = n >> 4;
  int lane = (n & 15) + 16 * (kk >> 4);
  int jj = kk & 15;
  out[((chunk * 4 + ntile) * 32 + lane) * 16 + jj] = (_Float16)acc;
}

__global__ void fold_b(const float* ba, const float* Wacc, const float* bacc, float* out) {
  int n = threadIdx.x;
  if (n >= 64) return;
  float acc = bacc[n];
#pragma unroll 8
  for (int j = 0; j < 64; ++j) acc += ba[j] * Wacc[j * 64 + n];
  out[n] = acc;
}

// ======================= generic WMMA linear (templated) =======================
// out[M x NOUT] = concat(s0[.,w0], s1[.,64], s2[.,64]) (padded to KPAD) @ Wp + bias
// 128 threads = 4 waves; wave w owns N-tiles [w*TILES, (w+1)*TILES).
template <int KPAD, int NOUT>
__global__ void linear_wmma(const float* __restrict__ s0, int w0,
                            const float* __restrict__ s1, int w1,
                            const float* __restrict__ s2, int w2,
                            const _Float16* __restrict__ Wp,
                            const float* __restrict__ bias,
                            float* __restrict__ out) {
  constexpr int CHUNKS = KPAD / 32;
  constexpr int TILESN = NOUT / 16;
  constexpr int TILES  = TILESN / 4;   // per wave
  __shared__ v16h AtV[CHUNKS * 32];    // frag-ordered A tile, 32B aligned
  _Float16* At = (_Float16*)AtV;

  const int row0 = blockIdx.x * 16;
  const int tid = threadIdx.x, lane = tid & 31, wv = tid >> 5;

  for (int i = tid; i < 16 * KPAD; i += 128) {
    int r = i / KPAD, c = i % KPAD;
    int gr = row0 + r;
    float v = 0.f;
    if (c < w0)                v = s0[(size_t)gr * w0 + c];
    else if (c < w0 + w1)      v = s1[(size_t)gr * 64 + (c - w0)];
    else if (c < w0 + w1 + w2) v = s2[(size_t)gr * 64 + (c - w0 - w1)];
    At[afrag_off(r, c)] = (_Float16)v;
  }
  __syncthreads();

  v8f acc[TILES];
#pragma unroll
  for (int t = 0; t < TILES; ++t)
#pragma unroll
    for (int j = 0; j < 8; ++j) acc[t][j] = 0.f;

#pragma unroll
  for (int kc = 0; kc < CHUNKS; ++kc) {
    v16h a = AtV[kc * 32 + lane];
#pragma unroll
    for (int t = 0; t < TILES; ++t) {
      int ntile = wv * TILES + t;
      v16h b = *(const v16h*)(Wp + ((size_t)(kc * TILESN + ntile) * 32 + lane) * 16);
      acc[t] = __builtin_amdgcn_wmma_f32_16x16x32_f16(false, a, false, b,
                                                      (short)0, acc[t], false, false);
    }
  }

  int nlo = lane & 15, half = lane >> 4;
#pragma unroll
  for (int t = 0; t < TILES; ++t) {
    int n = (wv * TILES + t) * 16 + nlo;
#pragma unroll
    for (int r = 0; r < 8; ++r) {
      int m = r + half * 8;
      out[(size_t)(row0 + m) * NOUT + n] = acc[t][r] + bias[n];
    }
  }
}

// ======================= edge attention (TransformerConv) =======================
// qkvs layout per node: [q(64) | k(64) | v(64) | s(64)]
__global__ void edge_alpha(const float* __restrict__ qkvs, const int* __restrict__ src,
                           const int* __restrict__ dst, int E,
                           float* __restrict__ alpha, float* __restrict__ nmax) {
  int e = blockIdx.x * blockDim.x + threadIdx.x;
  if (e >= E) return;
  int s = src[e], d = dst[e];
  const float* q = qkvs + (size_t)d * 256;
  const float* k = qkvs + (size_t)s * 256 + 64;
  float acc = 0.f;
#pragma unroll 8
  for (int c = 0; c < 64; ++c) acc += q[c] * k[c];
  acc *= 0.125f;  // 1/sqrt(64)
  alpha[e] = acc;
  atomicMaxF(&nmax[d], acc);
}

__global__ void edge_exp(const int* __restrict__ dst, int E,
                         float* __restrict__ alpha, const float* __restrict__ nmax,
                         float* __restrict__ nsum) {
  int e = blockIdx.x * blockDim.x + threadIdx.x;
  if (e >= E) return;
  int d = dst[e];
  float v = expf(alpha[e] - nmax[d]);
  alpha[e] = v;
  atomicAdd(&nsum[d], v);
}

__global__ void edge_agg(const float* __restrict__ qkvs, const int* __restrict__ src,
                         const int* __restrict__ dst, int E,
                         const float* __restrict__ alpha, const float* __restrict__ nsum,
                         float* __restrict__ outagg) {
  int e = blockIdx.x * blockDim.x + threadIdx.x;
  if (e >= E) return;
  int s = src[e], d = dst[e];
  float a = alpha[e] / nsum[d];
  const float* v = qkvs + (size_t)s * 256 + 128;
  float* o = outagg + (size_t)d * 64;
#pragma unroll 8
  for (int c = 0; c < 64; ++c) atomicAdd(&o[c], a * v[c]);
}

// beta-gated combine + relu:  x = relu(beta*xr + (1-beta)*out)
__global__ void beta_relu(const float* __restrict__ outagg, const float* __restrict__ qkvs,
                          const float* __restrict__ Wb, float* __restrict__ xout, int M) {
  int n = blockIdx.x * blockDim.x + threadIdx.x;
  if (n >= M) return;
  const float* o = outagg + (size_t)n * 64;
  const float* r = qkvs + (size_t)n * 256 + 192;  // s-projection
  float z = 0.f;
#pragma unroll 8
  for (int c = 0; c < 64; ++c)
    z += o[c] * Wb[c] + r[c] * Wb[64 + c] + (o[c] - r[c]) * Wb[128 + c];
  float beta = 1.f / (1.f + expf(-z));
  float* x = xout + (size_t)n * 64;
#pragma unroll 8
  for (int c = 0; c < 64; ++c)
    x[c] = fmaxf(beta * r[c] + (1.f - beta) * o[c], 0.f);
}

// GraphNorm (in-place): per (graph, channel)
__global__ void gnorm(float* __restrict__ x, const float* __restrict__ w,
                      const float* __restrict__ b, const float* __restrict__ ms) {
  int g = blockIdx.x, c = threadIdx.x;  // 64 threads
  size_t base = (size_t)g * NPG * 64 + c;
  float s = 0.f;
  for (int n = 0; n < NPG; ++n) s += x[base + (size_t)n * 64];
  float shift = (s * (1.f / NPG)) * ms[c];
  float var = 0.f;
  for (int n = 0; n < NPG; ++n) {
    float d = x[base + (size_t)n * 64] - shift;
    var += d * d;
  }
  float scale = w[c] * rsqrtf(var * (1.f / NPG) + 1e-5f);
  for (int n = 0; n < NPG; ++n) {
    float d = x[base + (size_t)n * 64] - shift;
    x[base + (size_t)n * 64] = scale * d + b[c];
  }
}

// gate logit per node: concat(x1, xa, xb) @ Wg + bg
__global__ void gate_logit(const float* __restrict__ x1, const float* __restrict__ xa,
                           const float* __restrict__ xb, const float* __restrict__ Wg,
                           const float* __restrict__ bg, float* __restrict__ gl, int M) {
  int n = blockIdx.x * blockDim.x + threadIdx.x;
  if (n >= M) return;
  float acc = bg[0];
#pragma unroll
  for (int i = 0; i < 15; ++i) acc += x1[(size_t)n * 15 + i] * Wg[i];
#pragma unroll 8
  for (int i = 0; i < 64; ++i) acc += xa[(size_t)n * 64 + i] * Wg[15 + i];
#pragma unroll 8
  for (int i = 0; i < 64; ++i) acc += xb[(size_t)n * 64 + i] * Wg[79 + i];
  gl[n] = acc;
}

// per-graph softmax-gated pooling + head -> V
__global__ void pool_head(const float* __restrict__ gl, const float* __restrict__ h,
                          const float* __restrict__ x2, const float* __restrict__ Wl1,
                          const float* __restrict__ bl1, float* __restrict__ Vout) {
  int g = blockIdx.x, c = threadIdx.x;  // 64 threads
  __shared__ float sg[NPG];
  __shared__ float sp[64];
  __shared__ float ssum;
  sg[c] = gl[g * NPG + c];
  sg[c + 64] = gl[g * NPG + 64 + c];
  __syncthreads();
  if (c == 0) {
    float m = -1e30f;
    for (int n = 0; n < NPG; ++n) m = fmaxf(m, sg[n]);
    float s = 0.f;
    for (int n = 0; n < NPG; ++n) { sg[n] = expf(sg[n] - m); s += sg[n]; }
    ssum = s;
  }
  __syncthreads();
  float acc = 0.f;
  for (int n = 0; n < NPG; ++n)
    acc += sg[n] * h[((size_t)g * NPG + n) * 64 + c];
  sp[c] = fmaxf(acc / ssum, 0.f);
  __syncthreads();
  if (c == 0) {
    float xg = bl1[0];
#pragma unroll 8
    for (int i = 0; i < 64; ++i) xg += sp[i] * Wl1[i];
#pragma unroll
    for (int i = 0; i < 4; ++i) xg += x2[g * 4 + i] * Wl1[64 + i];
    Vout[g] = tanhf(xg);
  }
}

// x_cat = concat(xa, xb) as f16 [NT x 128]
__global__ void xcat_pack(const float* __restrict__ xa, const float* __restrict__ xb,
                          _Float16* __restrict__ xc, int M) {
  int idx = blockIdx.x * blockDim.x + threadIdx.x;
  if (idx >= M * 128) return;
  int n = idx >> 7, c = idx & 127;
  float v = (c < 64) ? xa[(size_t)n * 64 + c] : xb[(size_t)n * 64 + (c - 64)];
  xc[idx] = (_Float16)v;
}

// ======================= orders gather-GEMM (folded through Wacc) =======================
// row = concat(data[i] (wd), xcat[idx0[i]] (128), [xcat[idx1[i]] (128)]) padded to KPAD
// tmp[seg[i]] += row @ Wc + bc     (NOUT = 64, tilesN = 4, one tile per wave)
template <int KPAD, bool HAS2>
__global__ void orders_wmma(const float* __restrict__ data, int wd,
                            const int* __restrict__ i0, const int* __restrict__ i1,
                            const _Float16* __restrict__ xc,
                            const _Float16* __restrict__ Wc, const float* __restrict__ bc,
                            const int* __restrict__ seg, float* __restrict__ tmp, int M) {
  constexpr int CHUNKS = KPAD / 32;
  __shared__ v16h AtV[CHUNKS * 32];
  _Float16* At = (_Float16*)AtV;

  const int row0 = blockIdx.x * 16;
  const int tid = threadIdx.x, lane = tid & 31, wv = tid >> 5;

  for (int i = tid; i < 16 * KPAD; i += 128) {
    int r = i / KPAD, c = i % KPAD;
    int gr = row0 + r;
    _Float16 v = (_Float16)0.f;
    if (gr < M) {
      if (c < wd)                  v = (_Float16)data[(size_t)gr * wd + c];
      else if (c < wd + 128)       v = xc[(size_t)i0[gr] * 128 + (c - wd)];
      else if (HAS2 && c < wd + 256) v = xc[(size_t)i1[gr] * 128 + (c - wd - 128)];
    }
    At[afrag_off(r, c)] = v;
  }
  __syncthreads();

  v8f acc;
#pragma unroll
  for (int j = 0; j < 8; ++j) acc[j] = 0.f;

#pragma unroll
  for (int kc = 0; kc < CHUNKS; ++kc) {
    v16h a = AtV[kc * 32 + lane];
    v16h b = *(const v16h*)(Wc + ((size_t)(kc * 4 + wv) * 32 + lane) * 16);
    acc = __builtin_amdgcn_wmma_f32_16x16x32_f16(false, a, false, b,
                                                 (short)0, acc, false, false);
  }

  int nlo = lane & 15, half = lane >> 4;
  int n = wv * 16 + nlo;
#pragma unroll
  for (int r = 0; r < 8; ++r) {
    int m = r + half * 8;
    int gr = row0 + m;
    if (gr < M) {
      int s = seg[gr];
      atomicAdd(&tmp[(size_t)s * 64 + n], acc[r] + bc[n]);
    }
  }
}

// p[s] = relu(tmp[s]) @ Wfin + bfin
__global__ void final_p(const float* __restrict__ tmp, const float* __restrict__ Wfin,
                        const float* __restrict__ bfin, float* __restrict__ p, int S) {
  int s = blockIdx.x * blockDim.x + threadIdx.x;
  if (s >= S) return;
  float acc = bfin[0];
#pragma unroll 8
  for (int c = 0; c < 64; ++c) acc += fmaxf(tmp[(size_t)s * 64 + c], 0.f) * Wfin[c];
  p[s] = acc;
}

__global__ void log_softmax16(const float* __restrict__ p, float* __restrict__ out) {
  int g = blockIdx.x * blockDim.x + threadIdx.x;
  if (g >= NB) return;
  const float* row = p + g * NMOV;
  float m = -1e30f;
#pragma unroll
  for (int i = 0; i < NMOV; ++i) m = fmaxf(m, row[i]);
  float s = 0.f;
#pragma unroll
  for (int i = 0; i < NMOV; ++i) s += expf(row[i] - m);
  float lse = logf(s);
#pragma unroll
  for (int i = 0; i < NMOV; ++i) out[g * NMOV + i] = row[i] - m - lse;
}

// ======================= host orchestration =======================
extern "C" void kernel_launch(void* const* d_in, const int* in_sizes, int n_in,
                              void* d_out, int out_size, void* d_ws, size_t ws_size,
                              hipStream_t stream) {
  // ---- inputs (setup_inputs dict flatten order) ----
  const float* x1     = (const float*)d_in[0];
  const float* x2     = (const float*)d_in[1];
  const int*   ei     = (const int*)d_in[2];      // [2, EDGES]
  const float* adata  = (const float*)d_in[4];
  const int*   asrcs  = (const int*)d_in[5];
  const int*   adsts  = (const int*)d_in[6];
  const float* tdata  = (const float*)d_in[7];
  const int*   tsrcs  = (const int*)d_in[8];
  const int*   tdsts  = (const int*)d_in[9];
  const float* ddata  = (const float*)d_in[10];
  const int*   dtgts  = (const int*)d_in[11];
  const int*   abtch  = (const int*)d_in[12];
  const int*   tbtch  = (const int*)d_in[13];
  const int*   dbtch  = (const int*)d_in[14];
  // params: g1 (16..24), g2 (25..33), norm1 (34..36), norm2 (37..39), heads (40..55)
  const float *Wq1=(const float*)d_in[16], *bq1=(const float*)d_in[17];
  const float *Wk1=(const float*)d_in[18], *bk1=(const float*)d_in[19];
  const float *Wv1=(const float*)d_in[20], *bv1=(const float*)d_in[21];
  const float *Ws1=(const float*)d_in[22], *bs1=(const float*)d_in[23];
  const float *Wb1=(const float*)d_in[24];
  const float *Wq2=(const float*)d_in[25], *bq2=(const float*)d_in[26];
  const float *Wk2=(const float*)d_in[27], *bk2=(const float*)d_in[28];
  const float *Wv2=(const float*)d_in[29], *bv2=(const float*)d_in[30];
  const float *Ws2=(const float*)d_in[31], *bs2=(const float*)d_in[32];
  const float *Wb2=(const float*)d_in[33];
  const float *n1w=(const float*)d_in[34], *n1b=(const float*)d_in[35], *n1ms=(const float*)d_in[36];
  const float *n2w=(const float*)d_in[37], *n2b=(const float*)d_in[38], *n2ms=(const float*)d_in[39];
  const float *Wg=(const float*)d_in[40], *bg=(const float*)d_in[41];
  const float *Wn=(const float*)d_in[42], *bn=(const float*)d_in[43];
  const float *Wl1=(const float*)d_in[44], *bl1=(const float*)d_in[45];
  const float *Wa=(const float*)d_in[46], *ba=(const float*)d_in[47];
  const float *Wt=(const float*)d_in[48], *bt=(const float*)d_in[49];
  const float *Wd=(const float*)d_in[50], *bd=(const float*)d_in[51];
  const float *Wacc=(const float*)d_in[52], *bacc=(const float*)d_in[53];
  const float *Wfin=(const float*)d_in[54], *bfin=(const float*)d_in[55];

  const int* esrc = ei;
  const int* edst = ei + EDGES;

  // ---- workspace layout ----
  size_t off = 0;
  auto alloc = [&](size_t bytes) -> void* {
    off = (off + 255) & ~(size_t)255;
    void* p = (char*)d_ws + off;
    off += bytes;
    return p;
  };
  float*     qkvs   = (float*)alloc((size_t)NTOT * 256 * 4);
  float*     alpha  = (float*)alloc((size_t)EDGES * 4);
  float*     nmax   = (float*)alloc((size_t)NTOT * 4);
  float*     nsum   = (float*)alloc((size_t)NTOT * 4);
  float*     outagg = (float*)alloc((size_t)NTOT * 64 * 4);
  float*     xa     = (float*)alloc((size_t)NTOT * 64 * 4);
  float*     xb     = (float*)alloc((size_t)NTOT * 64 * 4);
  float*     hbuf   = (float*)alloc((size_t)NTOT * 64 * 4);
  float*     gl     = (float*)alloc((size_t)NTOT * 4);
  _Float16*  xcat   = (_Float16*)alloc((size_t)NTOT * 128 * 2);
  float*     tmp    = (float*)alloc((size_t)NSEG * 64 * 4);
  float*     pbuf   = (float*)alloc((size_t)NSEG * 4);
  _Float16*  Wp1    = (_Float16*)alloc(32  * 256 * 2);
  _Float16*  Wp2    = (_Float16*)alloc(96  * 256 * 2);
  _Float16*  Wnp    = (_Float16*)alloc(160 * 64 * 2);
  _Float16*  Wap    = (_Float16*)alloc(320 * 64 * 2);
  _Float16*  Wtp    = (_Float16*)alloc(320 * 64 * 2);
  _Float16*  Wdp    = (_Float16*)alloc(160 * 64 * 2);
  float*     bp1    = (float*)alloc(256 * 4);
  float*     bp2    = (float*)alloc(256 * 4);
  float*     bap    = (float*)alloc(64 * 4);
  float*     btp    = (float*)alloc(64 * 4);
  float*     bdp    = (float*)alloc(64 * 4);

  float* Vout   = (float*)d_out;         // V[512]
  float* lsmOut = (float*)d_out + NB;    // logp[512*16]

  // ---- weight preparation (deterministic, small) ----
  pack_w4<<<(32 * 256 + 255) / 256, 256, 0, stream>>>(Wq1, Wk1, Wv1, Ws1, 15, 32, Wp1);
  pack_b4<<<1, 256, 0, stream>>>(bq1, bk1, bv1, bs1, bp1);
  pack_w4<<<(96 * 256 + 255) / 256, 256, 0, stream>>>(Wq2, Wk2, Wv2, Ws2, 79, 96, Wp2);
  pack_b4<<<1, 256, 0, stream>>>(bq2, bk2, bv2, bs2, bp2);
  pack_w1<<<(160 * 64 + 255) / 256, 256, 0, stream>>>(Wn, 143, 160, Wnp);
  fold_w<<<(320 * 64 + 255) / 256, 256, 0, stream>>>(Wa, Wacc, 308, 320, Wap);
  fold_w<<<(320 * 64 + 255) / 256, 256, 0, stream>>>(Wt, Wacc, 307, 320, Wtp);
  fold_w<<<(160 * 64 + 255) / 256, 256, 0, stream>>>(Wd, Wacc, 154, 160, Wdp);
  fold_b<<<1, 64, 0, stream>>>(ba, Wacc, bacc, bap);
  fold_b<<<1, 64, 0, stream>>>(bt, Wacc, bacc, btp);
  fold_b<<<1, 64, 0, stream>>>(bd, Wacc, bacc, bdp);

  // ---- layer 1 ----
  fill_f32<<<(NTOT + 255) / 256, 256, 0, stream>>>(nmax, NTOT, -1e30f);
  fill_f32<<<(NTOT + 255) / 256, 256, 0, stream>>>(nsum, NTOT, 0.f);
  fill_f32<<<(NTOT * 64 + 255) / 256, 256, 0, stream>>>(outagg, NTOT * 64, 0.f);

  linear_wmma<32, 256><<<NTOT / 16, 128, 0, stream>>>(x1, 15, nullptr, 0, nullptr, 0,
                                                      Wp1, bp1, qkvs);
  edge_alpha<<<EDGES / 256, 256, 0, stream>>>(qkvs, esrc, edst, EDGES, alpha, nmax);
  edge_exp<<<EDGES / 256, 256, 0, stream>>>(edst, EDGES, alpha, nmax, nsum);
  edge_agg<<<EDGES / 256, 256, 0, stream>>>(qkvs, esrc, edst, EDGES, alpha, nsum, outagg);
  beta_relu<<<NTOT / 256, 256, 0, stream>>>(outagg, qkvs, Wb1, xa, NTOT);
  gnorm<<<NB, 64, 0, stream>>>(xa, n1w, n1b, n1ms);

  // ---- layer 2 ----
  fill_f32<<<(NTOT + 255) / 256, 256, 0, stream>>>(nmax, NTOT, -1e30f);
  fill_f32<<<(NTOT + 255) / 256, 256, 0, stream>>>(nsum, NTOT, 0.f);
  fill_f32<<<(NTOT * 64 + 255) / 256, 256, 0, stream>>>(outagg, NTOT * 64, 0.f);

  linear_wmma<96, 256><<<NTOT / 16, 128, 0, stream>>>(x1, 15, xa, 64, nullptr, 0,
                                                      Wp2, bp2, qkvs);
  edge_alpha<<<EDGES / 256, 256, 0, stream>>>(qkvs, esrc, edst, EDGES, alpha, nmax);
  edge_exp<<<EDGES / 256, 256, 0, stream>>>(edst, EDGES, alpha, nmax, nsum);
  edge_agg<<<EDGES / 256, 256, 0, stream>>>(qkvs, esrc, edst, EDGES, alpha, nsum, outagg);
  beta_relu<<<NTOT / 256, 256, 0, stream>>>(outagg, qkvs, Wb2, xb, NTOT);
  gnorm<<<NB, 64, 0, stream>>>(xb, n2w, n2b, n2ms);

  // ---- pooling head -> V ----
  linear_wmma<160, 64><<<NTOT / 16, 128, 0, stream>>>(x1, 15, xa, 64, xb, 64,
                                                      Wnp, bn, hbuf);
  gate_logit<<<NTOT / 256, 256, 0, stream>>>(x1, xa, xb, Wg, bg, gl, NTOT);
  pool_head<<<NB, 64, 0, stream>>>(gl, hbuf, x2, Wl1, bl1, Vout);

  // ---- orders (folded through Wacc) ----
  xcat_pack<<<(NTOT * 128 + 255) / 256, 256, 0, stream>>>(xa, xb, xcat, NTOT);
  fill_f32<<<(NSEG * 64 + 255) / 256, 256, 0, stream>>>(tmp, NSEG * 64, 0.f);
  orders_wmma<320, true><<<AORD / 16, 128, 0, stream>>>(adata, 52, asrcs, adsts, xcat,
                                                        Wap, bap, abtch, tmp, AORD);
  orders_wmma<320, true><<<TORD / 16, 128, 0, stream>>>(tdata, 51, tsrcs, tdsts, xcat,
                                                        Wtp, btp, tbtch, tmp, TORD);
  orders_wmma<160, false><<<DORD / 16, 128, 0, stream>>>(ddata, 26, dtgts, nullptr, xcat,
                                                         Wdp, bdp, dbtch, tmp, DORD);

  final_p<<<NSEG / 256, 256, 0, stream>>>(tmp, Wfin, bfin, pbuf, NSEG);
  log_softmax16<<<(NB + 255) / 256, 256, 0, stream>>>(pbuf, lsmOut);

  (void)in_sizes; (void)n_in; (void)out_size; (void)ws_size;
}
